// TimeAwareMultiHeadAttention_77446850282054
// MI455X (gfx1250) — matrix operand order repaired and checked by reading, not compiled, not run on hardware
//
#include <hip/hip_runtime.h>
#include <math.h>

typedef __attribute__((ext_vector_type(16))) _Float16 v16h;
typedef __attribute__((ext_vector_type(8)))  float    v8f;
typedef __attribute__((ext_vector_type(4)))  float    v4f;
typedef __attribute__((ext_vector_type(2)))  float    v2f;

#define NB   32      // batch
#define SL   200     // sequence length
#define HD   64      // hidden
#define NHD  2       // heads
#define HSZ  32      // head size
#define HB   64      // NHD*NB
#define LL   40000   // SL*SL
#define NQT  13      // ceil(SL/16)
#define TPW  8       // row-tiles per wave in time_bias

static __device__ __forceinline__ int imin(int a, int b) { return a < b ? a : b; }

// ---------------------------------------------------------------------------
// Kernel 1: QKV projection + split-heads.  Qh/Kh/Vh layout: [hb, l, hs],
// hb = h*NB + b.  Tiny cost; plain VALU.
// ---------------------------------------------------------------------------
__global__ __launch_bounds__(256) void qkv_project_kernel(
    const float* __restrict__ qin, const float* __restrict__ kin,
    const float* __restrict__ vin,
    const float* __restrict__ Wq, const float* __restrict__ bq,
    const float* __restrict__ Wk, const float* __restrict__ bk,
    const float* __restrict__ Wv, const float* __restrict__ bv,
    float* __restrict__ Qh, float* __restrict__ Kh, float* __restrict__ Vh)
{
  int idx = blockIdx.x * blockDim.x + threadIdx.x;
  if (idx >= HB * SL * HSZ) return;
  int hs = idx % HSZ;
  int l  = (idx / HSZ) % SL;
  int hb = idx / (HSZ * SL);
  int b = hb % NB, h = hb / NB;
  int col = h * HSZ + hs;
  const float* qr = qin + (size_t)(b * SL + l) * HD;
  const float* kr = kin + (size_t)(b * SL + l) * HD;
  const float* vr = vin + (size_t)(b * SL + l) * HD;
  float aq = bq[col], ak = bk[col], av = bv[col];
#pragma unroll 8
  for (int c = 0; c < HD; ++c) {
    aq = fmaf(qr[c], Wq[c * HD + col], aq);
    ak = fmaf(kr[c], Wk[c * HD + col], ak);
    av = fmaf(vr[c], Wv[c * HD + col], av);
  }
  Qh[idx] = aq; Kh[idx] = ak; Vh[idx] = av;
}

// ---------------------------------------------------------------------------
// Kernel 2 (dominant, streams 655 MB): log(sigmoid(relu(T@Wt+bt)@Wp+bp))
// via v_wmma_f32_16x16x32_f16.  Rows = linear (hb,q,k) index; one wave does
// TPW tiles of 16 rows.  The read-once T stream is loaded NON-TEMPORAL so it
// does not thrash the 192MB L2 (keeps Wt / bias / QKV resident).  Result is
// written into the attn region of d_out (consumed as bias by kernel 3, then
// overwritten with the softmax output).
// ---------------------------------------------------------------------------
__global__ __launch_bounds__(256) void time_bias_kernel(
    const float* __restrict__ T,   // [HB, SL, SL, HD]
    const float* __restrict__ Wt, const float* __restrict__ bt,
    const float* __restrict__ Wp, const float* __restrict__ bp,
    float* __restrict__ biasOut)   // [HB*SL*SL]
{
  const int lane = threadIdx.x & 31;
  const int half = lane >> 4;
  const int lm   = lane & 15;
  const long long wave = (long long)((blockIdx.x * blockDim.x + threadIdx.x) >> 5);

  // B fragments of Wt (contiguous k order used here: lanes 0-15 K=0..15, lanes 16-31 K=16..31)
  v16h wfrag[4][2];
#pragma unroll
  for (int nt = 0; nt < 4; ++nt) {
    int n = nt * 16 + lm;
#pragma unroll
    for (int kc = 0; kc < 2; ++kc) {
      int kb = kc * 32 + half * 16;
#pragma unroll
      for (int j = 0; j < 16; ++j)
        wfrag[nt][kc][j] = (_Float16)Wt[(kb + j) * HD + n];
    }
  }
  float btn[4], wpn[4];
#pragma unroll
  for (int nt = 0; nt < 4; ++nt) { btn[nt] = bt[nt * 16 + lm]; wpn[nt] = Wp[nt * 16 + lm]; }
  const float bps = bp[0];

  const long long ntiles = (long long)HB * LL / 16;  // 160000
  long long tile0 = wave * TPW;
  for (int tt = 0; tt < TPW; ++tt) {
    long long tile = tile0 + tt;
    if (tile >= ntiles) break;
    long long row = tile * 16 + lm;                  // this lane's matrix row (M=lm)
    const v4f* ar4 = (const v4f*)(T + row * (long long)HD);  // 16 x v4f per row

    // A fragments (contiguous k order, same as B: lanes<16 hold K 0-15, lanes>=16 hold
    // K 16-31).  Non-temporal b128 loads: stream is read exactly once.
    v16h af[2];
#pragma unroll
    for (int kc = 0; kc < 2; ++kc) {
      int f0 = kc * 8 + half * 4;                    // float4 index of K-base
      v4f x0 = __builtin_nontemporal_load(ar4 + f0);
      v4f x1 = __builtin_nontemporal_load(ar4 + f0 + 1);
      v4f x2 = __builtin_nontemporal_load(ar4 + f0 + 2);
      v4f x3 = __builtin_nontemporal_load(ar4 + f0 + 3);
#pragma unroll
      for (int j = 0; j < 4; ++j) {
        af[kc][j]      = (_Float16)x0[j];
        af[kc][4 + j]  = (_Float16)x1[j];
        af[kc][8 + j]  = (_Float16)x2[j];
        af[kc][12 + j] = (_Float16)x3[j];
      }
    }

    v8f acc[4];
#pragma unroll
    for (int nt = 0; nt < 4; ++nt) {
      v8f a = {0.f, 0.f, 0.f, 0.f, 0.f, 0.f, 0.f, 0.f};
      a = __builtin_amdgcn_wmma_f32_16x16x32_f16(false, af[0], false, wfrag[nt][0],
                                                 (short)0, a, false, false);
      a = __builtin_amdgcn_wmma_f32_16x16x32_f16(false, af[1], false, wfrag[nt][1],
                                                 (short)0, a, false, false);
      acc[nt] = a;
    }

    // relu(hidden + bt) dotted with Wp: per-lane partial over its N column,
    // then sum across the 16 lanes of each half (rows m = r + 8*half).
    v8f s = {0.f, 0.f, 0.f, 0.f, 0.f, 0.f, 0.f, 0.f};
#pragma unroll
    for (int nt = 0; nt < 4; ++nt)
#pragma unroll
      for (int r = 0; r < 8; ++r) {
        float hv = fmaxf(acc[nt][r] + btn[nt], 0.f);
        s[r] = fmaf(hv, wpn[nt], s[r]);
      }
#pragma unroll
    for (int r = 0; r < 8; ++r) {
      float x = s[r];
      x += __shfl_xor(x, 1, 32);
      x += __shfl_xor(x, 2, 32);
      x += __shfl_xor(x, 4, 32);
      x += __shfl_xor(x, 8, 32);
      s[r] = x;
    }
    if (lm == 0) {  // lane 0 stores rows 0..7, lane 16 stores rows 8..15
      float o[8];
#pragma unroll
      for (int r = 0; r < 8; ++r) {
        float p = s[r] + bps;
        o[r] = -log1pf(__expf(-p));   // log(sigmoid(p))
      }
      long long base = tile * 16 + half * 8;
      *(float4*)(biasOut + base)     = make_float4(o[0], o[1], o[2], o[3]);
      *(float4*)(biasOut + base + 4) = make_float4(o[4], o[5], o[6], o[7]);
    }
  }
}

// ---------------------------------------------------------------------------
// Kernel 3: attention.  One wave per (hb, 16-query tile).
// scores = QK^T (wmma f32 16x16x4), + log-bias, /sqrt(32), mask, softmax,
// write attn, then out = attn @ V (wmma f32 16x16x4).
// ---------------------------------------------------------------------------
__global__ __launch_bounds__(256) void attention_kernel(
    const float* __restrict__ Qh, const float* __restrict__ Kh,
    const float* __restrict__ Vh, const int* __restrict__ tmask,
    float* __restrict__ attnOut, float* __restrict__ outp)
{
  const int lane = threadIdx.x & 31;
  const int half = lane >> 4;
  const int lm   = lane & 15;
  int wid = (blockIdx.x * blockDim.x + threadIdx.x) >> 5;
  if (wid >= HB * NQT) return;
  int hb = wid / NQT;
  int qt = wid % NQT;
  int qbase = qt * 16;
  int b = hb % NB;
  int h = hb / NB;

  // Q fragments (f32 A 16x4: lanes<16 K={4c,4c+1}, lanes>=16 K={4c+2,4c+3})
  int qclamp = imin(qbase + lm, SL - 1);
  const float* qp = Qh + ((size_t)hb * SL + qclamp) * HSZ;
  v2f qf[8];
#pragma unroll
  for (int c = 0; c < 8; ++c) {
    int k0 = c * 4 + half * 2;
    qf[c].x = qp[k0]; qf[c].y = qp[k0 + 1];
  }

  int qv[8], tmv[8];
#pragma unroll
  for (int r = 0; r < 8; ++r) {
    int q = qbase + r + 8 * half;
    qv[r] = q;
    tmv[r] = tmask[b * SL + imin(q, SL - 1)];
  }

  const float* biasBase = attnOut + (size_t)hb * LL;
  const float scale = 0.17677669529663687f;      // 1/sqrt(32)
  const float NEGC = -4294967295.0f;             // -2^32 + 1
  v8f sc[NQT];

#pragma unroll
  for (int t = 0; t < NQT; ++t) {
    int n = t * 16 + lm;
    int nc = imin(n, SL - 1);
    const float* kp = Kh + ((size_t)hb * SL + nc) * HSZ;
    v8f a = {0.f, 0.f, 0.f, 0.f, 0.f, 0.f, 0.f, 0.f};
#pragma unroll
    for (int c = 0; c < 8; ++c) {
      int k0 = c * 4 + half * 2;
      v2f bf; bf.x = kp[k0]; bf.y = kp[k0 + 1];
      a = __builtin_amdgcn_wmma_f32_16x16x4_f32(false, qf[c], false, bf,
                                                (short)0, a, false, false);
    }
    bool nvalid = n < SL;
#pragma unroll
    for (int r = 0; r < 8; ++r) {
      int q = qv[r];
      float bias = biasBase[(size_t)imin(q, SL - 1) * SL + nc];
      float v = (a[r] + bias) * scale;
      bool masked = (tmv[r] != 0) || (n > q);
      v = masked ? NEGC : v;
      a[r] = (nvalid && q < SL) ? v : -INFINITY;  // out-of-range excluded entirely
    }
    sc[t] = a;
  }

  // row max (across key tiles, then across the 16 lanes of each half)
  v8f vmax = sc[0];
#pragma unroll
  for (int t = 1; t < NQT; ++t)
#pragma unroll
    for (int r = 0; r < 8; ++r) vmax[r] = fmaxf(vmax[r], sc[t][r]);
#pragma unroll
  for (int r = 0; r < 8; ++r) {
    float x = vmax[r];
    x = fmaxf(x, __shfl_xor(x, 1, 32));
    x = fmaxf(x, __shfl_xor(x, 2, 32));
    x = fmaxf(x, __shfl_xor(x, 4, 32));
    x = fmaxf(x, __shfl_xor(x, 8, 32));
    vmax[r] = x;
  }
  v8f vsum = {0.f, 0.f, 0.f, 0.f, 0.f, 0.f, 0.f, 0.f};
#pragma unroll
  for (int t = 0; t < NQT; ++t) {
    int n = t * 16 + lm;
    bool nvalid = n < SL;
#pragma unroll
    for (int r = 0; r < 8; ++r) {
      float e = (nvalid && qv[r] < SL) ? __expf(sc[t][r] - vmax[r]) : 0.f;
      sc[t][r] = e;
      vsum[r] += e;
    }
  }
#pragma unroll
  for (int r = 0; r < 8; ++r) {
    float x = vsum[r];
    x += __shfl_xor(x, 1, 32);
    x += __shfl_xor(x, 2, 32);
    x += __shfl_xor(x, 4, 32);
    x += __shfl_xor(x, 8, 32);
    vsum[r] = x;
  }
  v8f inv;
#pragma unroll
  for (int r = 0; r < 8; ++r) inv[r] = 1.0f / vsum[r];

  // store attn (overwrites the bias this wave already consumed)
  float* attnW = attnOut + (size_t)hb * LL;
#pragma unroll
  for (int t = 0; t < NQT; ++t) {
    int n = t * 16 + lm;
    if (n < SL) {
#pragma unroll
      for (int r = 0; r < 8; ++r) {
        int q = qv[r];
        if (q < SL) attnW[(size_t)q * SL + n] = sc[t][r] * inv[r];
      }
    }
  }

  // out = attn @ V : K-dim = 200 keys = 50 chunks of 4, N = 32 head dims (2 tiles)
  const float* ap = attnW + (size_t)qclamp * SL;
  const float* vp = Vh + (size_t)hb * SL * HSZ;
  v8f oacc[2] = {{0.f,0.f,0.f,0.f,0.f,0.f,0.f,0.f}, {0.f,0.f,0.f,0.f,0.f,0.f,0.f,0.f}};
  for (int c = 0; c < 50; ++c) {
    int k0 = c * 4 + half * 2;
    v2f af2; af2.x = ap[k0]; af2.y = ap[k0 + 1];
#pragma unroll
    for (int nt = 0; nt < 2; ++nt) {
      int nhs = nt * 16 + lm;
      v2f bf2;
      bf2.x = vp[(size_t)k0 * HSZ + nhs];
      bf2.y = vp[(size_t)(k0 + 1) * HSZ + nhs];
      oacc[nt] = __builtin_amdgcn_wmma_f32_16x16x4_f32(false, af2, false, bf2,
                                                       (short)0, oacc[nt], false, false);
    }
  }
#pragma unroll
  for (int nt = 0; nt < 2; ++nt) {
    int nhs = nt * 16 + lm;
#pragma unroll
    for (int r = 0; r < 8; ++r) {
      int q = qv[r];
      if (q < SL)
        outp[((size_t)(b * SL + q)) * HD + h * HSZ + nhs] = oacc[nt][r];
    }
  }
}

// ---------------------------------------------------------------------------
extern "C" void kernel_launch(void* const* d_in, const int* in_sizes, int n_in,
                              void* d_out, int out_size, void* d_ws, size_t ws_size,
                              hipStream_t stream) {
  const float* queries = (const float*)d_in[0];
  const float* keys    = (const float*)d_in[1];
  const float* values  = (const float*)d_in[2];
  const float* tmat    = (const float*)d_in[3];
  const int*   tmask   = (const int*)d_in[4];
  // d_in[5] = attn_mask (causal) is computed analytically
  const float* Wq = (const float*)d_in[6];  const float* bq = (const float*)d_in[7];
  const float* Wk = (const float*)d_in[8];  const float* bk = (const float*)d_in[9];
  const float* Wv = (const float*)d_in[10]; const float* bv = (const float*)d_in[11];
  const float* Wt = (const float*)d_in[12]; const float* bt = (const float*)d_in[13];
  const float* Wp = (const float*)d_in[14]; const float* bp = (const float*)d_in[15];

  float* outp    = (float*)d_out;                 // [32,200,64]
  float* attnOut = outp + (size_t)NB * SL * HD;   // [64,200,200], also bias scratch

  float* Qh = (float*)d_ws;                       // [64,200,32] each
  float* Kh = Qh + (size_t)HB * SL * HSZ;
  float* Vh = Kh + (size_t)HB * SL * HSZ;

  // 1) QKV projection + split heads
  {
    int total = HB * SL * HSZ;                    // 409600
    qkv_project_kernel<<<(total + 255) / 256, 256, 0, stream>>>(
        queries, keys, values, Wq, bq, Wk, bk, Wv, bv, Qh, Kh, Vh);
  }
  // 2) time-bias MLP (dominant; streams 655 MB through f16 WMMA, NT loads)
  {
    long long waves = (long long)HB * LL / 16 / TPW;  // 20000
    int blocks = (int)((waves * 32 + 255) / 256);     // 2500
    time_bias_kernel<<<blocks, 256, 0, stream>>>(tmat, Wt, bt, Wp, bp, attnOut);
  }
  // 3) attention
  {
    int waves = HB * NQT;                         // 832
    int blocks = (waves * 32 + 255) / 256;        // 104
    attention_kernel<<<blocks, 256, 0, stream>>>(Qh, Kh, Vh, tmask, attnOut, outp);
  }
}
